// MultiHeadAttention_72146860638705
// MI455X (gfx1250) — matrix operand order repaired and checked by reading, not compiled, not run on hardware
//
#include <hip/hip_runtime.h>

#define B_  2
#define L_  2048
#define D_  1024
#define H_  16
#define HD_ 64
#define M_  (B_ * L_)   // 4096

typedef __bf16 bf16_t;
typedef __attribute__((ext_vector_type(16))) __bf16 v16bf;
typedef __attribute__((ext_vector_type(8)))  __bf16 v8bf;
typedef __attribute__((ext_vector_type(4)))  __bf16 v4bf;
typedef __attribute__((ext_vector_type(8)))  float  v8f;
typedef __attribute__((ext_vector_type(4)))  unsigned v4u;
typedef __attribute__((ext_vector_type(8)))  unsigned v8u;

// A-fragment (16-bit A 16x32): lane holds two 8-elem chunks at +0 and +16 elems.
static __device__ inline v16bf frag_gap16(const bf16_t* p) {
  union { v16bf v; v8bf h[2]; } u;
  u.h[0] = *(const v8bf*)(p);
  u.h[1] = *(const v8bf*)(p + 16);
  return u.v;
}
// B-fragment: lane holds two 8-elem chunks at +0 and +16 elems (column of B == row of W / K / V^T).
static __device__ inline v16bf frag_contig(const bf16_t* p) {
  union { v16bf v; v8bf h[2]; } u;
  u.h[0] = *(const v8bf*)(p);
  u.h[1] = *(const v8bf*)(p + 16);
  return u.v;
}
static __device__ inline v8f wmma_bf16(v16bf a, v16bf b, v8f c) {
  return __builtin_amdgcn_wmma_f32_16x16x32_bf16(false, a, false, b, (short)0, c,
                                                 false, false);
}

// Async global->LDS 16B per lane, ASYNCcnt-tracked. IOFFSET applies to BOTH the
// LDS and global addresses, so one base pair serves a whole per-thread chunk.
template <int OFF>
static __device__ inline void async_b128(const bf16_t* lds, const bf16_t* gp) {
  unsigned l = (unsigned)(unsigned long long)lds;
  unsigned long long g = (unsigned long long)gp;
  asm volatile("global_load_async_to_lds_b128 %0, %1, off offset:%2"
               :: "v"(l), "v"(g), "i"(OFF) : "memory");
}
static __device__ inline void wait_async0() {
  asm volatile("s_wait_asynccnt 0x0" ::: "memory");
}

// ---------------------------------------------------------------- convert ---
__global__ __launch_bounds__(256)
void cvt_f32_bf16(const float* __restrict__ src, bf16_t* __restrict__ dst, int n4) {
  int i = blockIdx.x * blockDim.x + threadIdx.x;
  if (i < n4) {
    float4 f = ((const float4*)src)[i];
    v4bf o;
    o.x = (bf16_t)f.x; o.y = (bf16_t)f.y; o.z = (bf16_t)f.z; o.w = (bf16_t)f.w;
    ((v4bf*)dst)[i] = o;
  }
}

// ------------------------------------------------------------- QKV GEMM -----
// C[m,n] = A[m,:] . W[n,:] + bias[n]; z selects Q/K/V; scatter to attention layouts.
// BK=64, double-buffered async-to-LDS staging, batched fragment loads.
__global__ __launch_bounds__(256)
void qkv_gemm(const bf16_t* __restrict__ A,
              const bf16_t* __restrict__ Wq, const bf16_t* __restrict__ Wk,
              const bf16_t* __restrict__ Wv,
              const float* __restrict__ bq, const float* __restrict__ bk,
              const float* __restrict__ bv,
              bf16_t* __restrict__ Qo, bf16_t* __restrict__ Ko,
              bf16_t* __restrict__ VTo) {
  const int z = blockIdx.z;
  const bf16_t* W   = (z == 0) ? Wq : (z == 1) ? Wk : Wv;
  const float* bias = (z == 0) ? bq : (z == 1) ? bk : bv;
  const int m0 = blockIdx.x * 128;
  const int n0 = blockIdx.y * 64;
  const int tid = threadIdx.x;
  const int lane = tid & 31, wid = tid >> 5;
  const int r16 = lane & 15, khalf = lane >> 4, kb16 = khalf * 8;

  __shared__ __align__(16) bf16_t As[2][128 * 72];  // 128 x 64, stride 72 (144B)
  __shared__ __align__(16) bf16_t Ws[2][64 * 72];   // 64 x 64

  const int arow = tid >> 1, acol = (tid & 1) * 32;  // 32 elems / thread
  const int wrow = tid >> 2, wcol = (tid & 3) * 16;  // 16 elems / thread

  auto stage = [&](int k0, int buf) {
    const bf16_t* Ag = A + (size_t)(m0 + arow) * D_ + k0 + acol;
    bf16_t* Al = &As[buf][arow * 72 + acol];
    async_b128<0>(Al, Ag);
    async_b128<16>(Al, Ag);
    async_b128<32>(Al, Ag);
    async_b128<48>(Al, Ag);
    const bf16_t* Wg = W + (size_t)(n0 + wrow) * D_ + k0 + wcol;
    bf16_t* Wl = &Ws[buf][wrow * 72 + wcol];
    async_b128<0>(Wl, Wg);
    async_b128<16>(Wl, Wg);
  };

  v8f acc[4] = {};
  stage(0, 0);
  int cur = 0;
  for (int k0 = 0; k0 < D_; k0 += 64) {
    wait_async0();
    __syncthreads();
    if (k0 + 64 < D_) stage(k0 + 64, cur ^ 1);
#pragma unroll
    for (int kk = 0; kk < 2; ++kk) {
      v16bf af = frag_gap16(&As[cur][(wid * 16 + r16) * 72 + kk * 32 + khalf * 8]);
      v16bf bfr[4];
#pragma unroll
      for (int c = 0; c < 4; ++c)
        bfr[c] = frag_contig(&Ws[cur][(c * 16 + r16) * 72 + kk * 32 + kb16]);
#pragma unroll
      for (int c = 0; c < 4; ++c)
        acc[c] = wmma_bf16(af, bfr[c], acc[c]);
    }
    cur ^= 1;
  }

#pragma unroll
  for (int c = 0; c < 4; ++c) {
#pragma unroll
    for (int r = 0; r < 8; ++r) {
      int m = m0 + wid * 16 + r + ((lane >> 4) << 3);
      int n = n0 + c * 16 + r16;
      float v = acc[c][r] + bias[n];
      int bb = m >> 11, lq = m & (L_ - 1);
      int hh = n >> 6,  hd = n & 63;
      if (z == 2) {
        VTo[((size_t)(bb * H_ + hh) * HD_ + hd) * L_ + lq] = (bf16_t)v;
      } else {
        bf16_t* dst = (z == 0) ? Qo : Ko;
        dst[((size_t)(bb * H_ + hh) * L_ + lq) * HD_ + hd] = (bf16_t)v;
      }
    }
  }
}

// --------------------------------------------------------- flash attention --
__global__ __launch_bounds__(256)
void flash_attn(const bf16_t* __restrict__ Q, const bf16_t* __restrict__ K,
                const bf16_t* __restrict__ VT, bf16_t* __restrict__ AO) {
  const int h = blockIdx.y, b = blockIdx.z;
  const int q0 = blockIdx.x * 128;
  const int tid = threadIdx.x, lane = tid & 31, wid = tid >> 5;
  const int r16 = lane & 15, khalf = lane >> 4, kb16 = khalf * 8;
  const float scale = 0.125f;  // HD^-0.5

  __shared__ __align__(16) bf16_t Qs[128 * 72];
  __shared__ __align__(16) bf16_t Ps[128 * 72];
  __shared__ __align__(16) bf16_t Ks[2][64 * 72];
  __shared__ __align__(16) bf16_t VTs[2][64 * 72];

  const bf16_t* Qg  = Q  + ((size_t)(b * H_ + h) * L_ + q0) * HD_;
  const bf16_t* Kg  = K  + (size_t)(b * H_ + h) * L_ * HD_;
  const bf16_t* VTg = VT + (size_t)(b * H_ + h) * HD_ * L_;

  const int krow = tid >> 2, kcol = (tid & 3) * 16;  // K/VT tiles: 16 elems/thread

  auto stage_kv = [&](int kt, int buf) {
    const bf16_t* kg = Kg + (size_t)(kt + krow) * HD_ + kcol;
    bf16_t* kl = &Ks[buf][krow * 72 + kcol];
    async_b128<0>(kl, kg);
    async_b128<16>(kl, kg);
    const bf16_t* vg = VTg + (size_t)krow * L_ + kt + kcol;
    bf16_t* vl = &VTs[buf][krow * 72 + kcol];
    async_b128<0>(vl, vg);
    async_b128<16>(vl, vg);
  };

  // Stage Q tile (128 x 64 -> LDS rows padded to 144B) via the Tensor Data
  // Mover: one descriptor-driven DMA, issued by wave 0, tracked by TENSORcnt.
  if (wid == 0) {
    unsigned lds_addr = (unsigned)(unsigned long long)(&Qs[0]);
    unsigned long long ga = (unsigned long long)Qg;
    v4u g0;
    g0[0] = 1u;                                   // count=1, user descriptor
    g0[1] = lds_addr;                             // LDS byte address
    g0[2] = (unsigned)ga;                         // global_addr[31:0]
    g0[3] = ((unsigned)(ga >> 32) & 0x01FFFFFFu)  // global_addr[56:32]
            | 0x80000000u;                        // type = 2 ("image")
    v8u g1;
    // data_size=2B; pad_enable; pad_interval=4 (every 32 DWORDs = one 128B row);
    // pad_amount=3 (4 DWORDs = 16B) -> LDS row stride 144B.
    g1[0] = (1u << 16) | (1u << 20) | (4u << 22) | (3u << 25);
    g1[1] = ((unsigned)HD_ & 0xffffu) << 16;      // tensor_dim0 = 64 (bits 79:48)
    g1[2] = ((unsigned)L_ & 0xffffu) << 16;       // tensor_dim1 = 2048 (bits 111:80)
    g1[3] = ((unsigned)HD_) << 16;                // tile_dim0 = 64 (bits 127:112)
    g1[4] = 128u;                                 // tile_dim1 = 128; tile_dim2 = 0
    g1[5] = (unsigned)HD_;                        // tensor_dim0_stride = 64
    g1[6] = 0u;
    g1[7] = 0u;
    asm volatile("tensor_load_to_lds %0, %1" :: "s"(g0), "s"(g1) : "memory");
  }
  stage_kv(0, 0);
  if (wid == 0) __builtin_amdgcn_s_wait_tensorcnt((short)0);
  wait_async0();
  __syncthreads();

  // Q A-fragments are loop-invariant per wave: keep them in VGPRs for all tiles.
  v16bf aq[2];
#pragma unroll
  for (int kk = 0; kk < 2; ++kk)
    aq[kk] = frag_gap16(&Qs[(wid * 16 + r16) * 72 + kk * 32 + khalf * 8]);

  v8f o[4] = {};
  float m8[8], l8[8];
#pragma unroll
  for (int r = 0; r < 8; ++r) { m8[r] = -1e30f; l8[r] = 0.f; }

  int cur = 0;
  for (int kt = 0; kt < L_; kt += 64) {
    if (kt + 64 < L_) stage_kv(kt + 64, cur ^ 1);

    // S = Q . K^T  (16 x 64 per wave), batched fragment loads
    v8f s[4] = {};
#pragma unroll
    for (int kk = 0; kk < 2; ++kk) {
      v16bf bfr[4];
#pragma unroll
      for (int c = 0; c < 4; ++c)
        bfr[c] = frag_contig(&Ks[cur][(c * 16 + r16) * 72 + kk * 32 + kb16]);
#pragma unroll
      for (int c = 0; c < 4; ++c)
        s[c] = wmma_bf16(aq[kk], bfr[c], s[c]);
    }

    // online softmax (row stats per 16-lane group, 8 rows per lane)
#pragma unroll
    for (int r = 0; r < 8; ++r) {
      float mx = fmaxf(fmaxf(s[0][r], s[1][r]), fmaxf(s[2][r], s[3][r])) * scale;
      mx = fmaxf(mx, __shfl_xor(mx, 1));
      mx = fmaxf(mx, __shfl_xor(mx, 2));
      mx = fmaxf(mx, __shfl_xor(mx, 4));
      mx = fmaxf(mx, __shfl_xor(mx, 8));
      float mnew  = fmaxf(m8[r], mx);
      float alpha = __expf(m8[r] - mnew);
      m8[r] = mnew;
      l8[r] *= alpha;
      o[0][r] *= alpha; o[1][r] *= alpha; o[2][r] *= alpha; o[3][r] *= alpha;
      float rs = 0.f;
#pragma unroll
      for (int c = 0; c < 4; ++c) {
        float p = __expf(s[c][r] * scale - mnew);
        s[c][r] = p;
        rs += p;
      }
      rs += __shfl_xor(rs, 1);
      rs += __shfl_xor(rs, 2);
      rs += __shfl_xor(rs, 4);
      rs += __shfl_xor(rs, 8);
      l8[r] += rs;
    }

    // P: C-fragment layout -> LDS -> A-fragment layout (wave-private rows)
#pragma unroll
    for (int c = 0; c < 4; ++c)
#pragma unroll
      for (int r = 0; r < 8; ++r)
        Ps[(wid * 16 + r + ((lane >> 4) << 3)) * 72 + c * 16 + r16] = (bf16_t)s[c][r];

    // O += P . V  (V^T staged => contiguous B-fragment reads)
#pragma unroll
    for (int kk = 0; kk < 2; ++kk) {
      v16bf ap = frag_gap16(&Ps[(wid * 16 + r16) * 72 + kk * 32 + khalf * 8]);
      v16bf vfr[4];
#pragma unroll
      for (int c = 0; c < 4; ++c)
        vfr[c] = frag_contig(&VTs[cur][(c * 16 + r16) * 72 + kk * 32 + kb16]);
#pragma unroll
      for (int c = 0; c < 4; ++c)
        o[c] = wmma_bf16(ap, vfr[c], o[c]);
    }

    cur ^= 1;
    wait_async0();     // retire next tile's loads issued above
    __syncthreads();   // all waves done computing before buffers swap roles
  }

#pragma unroll
  for (int r = 0; r < 8; ++r) {
    float inv = 1.0f / l8[r];
    int lq = q0 + wid * 16 + r + ((lane >> 4) << 3);
#pragma unroll
    for (int c = 0; c < 4; ++c) {
      int n = h * HD_ + c * 16 + r16;
      AO[((size_t)(b * L_ + lq)) * D_ + n] = (bf16_t)(o[c][r] * inv);
    }
  }
}

// --------------------------------------------------------- out projection ---
__global__ __launch_bounds__(256)
void out_proj(const bf16_t* __restrict__ A, const bf16_t* __restrict__ W,
              const float* __restrict__ bias, float* __restrict__ out) {
  const int m0 = blockIdx.x * 128;
  const int n0 = blockIdx.y * 64;
  const int tid = threadIdx.x;
  const int lane = tid & 31, wid = tid >> 5;
  const int r16 = lane & 15, khalf = lane >> 4, kb16 = khalf * 8;

  __shared__ __align__(16) bf16_t As[2][128 * 72];
  __shared__ __align__(16) bf16_t Ws[2][64 * 72];

  const int arow = tid >> 1, acol = (tid & 1) * 32;
  const int wrow = tid >> 2, wcol = (tid & 3) * 16;

  auto stage = [&](int k0, int buf) {
    const bf16_t* Ag = A + (size_t)(m0 + arow) * D_ + k0 + acol;
    bf16_t* Al = &As[buf][arow * 72 + acol];
    async_b128<0>(Al, Ag);
    async_b128<16>(Al, Ag);
    async_b128<32>(Al, Ag);
    async_b128<48>(Al, Ag);
    const bf16_t* Wg = W + (size_t)(n0 + wrow) * D_ + k0 + wcol;
    bf16_t* Wl = &Ws[buf][wrow * 72 + wcol];
    async_b128<0>(Wl, Wg);
    async_b128<16>(Wl, Wg);
  };

  v8f acc[4] = {};
  stage(0, 0);
  int cur = 0;
  for (int k0 = 0; k0 < D_; k0 += 64) {
    wait_async0();
    __syncthreads();
    if (k0 + 64 < D_) stage(k0 + 64, cur ^ 1);
#pragma unroll
    for (int kk = 0; kk < 2; ++kk) {
      v16bf af = frag_gap16(&As[cur][(wid * 16 + r16) * 72 + kk * 32 + khalf * 8]);
      v16bf bfr[4];
#pragma unroll
      for (int c = 0; c < 4; ++c)
        bfr[c] = frag_contig(&Ws[cur][(c * 16 + r16) * 72 + kk * 32 + kb16]);
#pragma unroll
      for (int c = 0; c < 4; ++c)
        acc[c] = wmma_bf16(af, bfr[c], acc[c]);
    }
    cur ^= 1;
  }

#pragma unroll
  for (int c = 0; c < 4; ++c) {
#pragma unroll
    for (int r = 0; r < 8; ++r) {
      int m = m0 + wid * 16 + r + ((lane >> 4) << 3);
      int n = n0 + c * 16 + r16;
      out[(size_t)m * D_ + n] = acc[c][r] + bias[n];
    }
  }
}

// ------------------------------------------------------------------ host ----
extern "C" void kernel_launch(void* const* d_in, const int* in_sizes, int n_in,
                              void* d_out, int out_size, void* d_ws, size_t ws_size,
                              hipStream_t stream) {
  (void)in_sizes; (void)n_in; (void)out_size; (void)ws_size;
  const float* query = (const float*)d_in[0];
  const float* Wq = (const float*)d_in[1];
  const float* bq = (const float*)d_in[2];
  const float* Wk = (const float*)d_in[3];
  const float* bk = (const float*)d_in[4];
  const float* Wv = (const float*)d_in[5];
  const float* bv = (const float*)d_in[6];
  const float* Wo = (const float*)d_in[7];
  const float* bo = (const float*)d_in[8];
  float* out = (float*)d_out;

  char* ws = (char*)d_ws;
  const size_t MLD = (size_t)M_ * D_;  // 4,194,304 elems
  const size_t DD  = (size_t)D_ * D_;  // 1,048,576 elems
  bf16_t* qbf = (bf16_t*)ws; ws += MLD * 2;
  bf16_t* wqb = (bf16_t*)ws; ws += DD * 2;
  bf16_t* wkb = (bf16_t*)ws; ws += DD * 2;
  bf16_t* wvb = (bf16_t*)ws; ws += DD * 2;
  bf16_t* wob = (bf16_t*)ws; ws += DD * 2;
  bf16_t* Qb  = (bf16_t*)ws; ws += MLD * 2;
  bf16_t* Kb  = (bf16_t*)ws; ws += MLD * 2;
  bf16_t* VTb = (bf16_t*)ws; ws += MLD * 2;
  bf16_t* AOb = (bf16_t*)ws; ws += MLD * 2;

  const int n4q = (int)(MLD / 4), n4w = (int)(DD / 4);
  cvt_f32_bf16<<<n4q / 256, 256, 0, stream>>>(query, qbf, n4q);
  cvt_f32_bf16<<<n4w / 256, 256, 0, stream>>>(Wq, wqb, n4w);
  cvt_f32_bf16<<<n4w / 256, 256, 0, stream>>>(Wk, wkb, n4w);
  cvt_f32_bf16<<<n4w / 256, 256, 0, stream>>>(Wv, wvb, n4w);
  cvt_f32_bf16<<<n4w / 256, 256, 0, stream>>>(Wo, wob, n4w);

  dim3 g1(M_ / 128, D_ / 64, 3);
  qkv_gemm<<<g1, 256, 0, stream>>>(qbf, wqb, wkb, wvb, bq, bk, bv, Qb, Kb, VTb);

  dim3 g2(L_ / 128, H_, B_);
  flash_attn<<<g2, 256, 0, stream>>>(Qb, Kb, VTb, AOb);

  dim3 g3(M_ / 128, D_ / 64, 1);
  out_proj<<<g3, 256, 0, stream>>>(AOb, wob, bo, out);
}
